// GPTLMHeadModel_21723944583722
// MI455X (gfx1250) — compile-verified, matrix operand order checked by reading, not run on hardware
//
#include <hip/hip_runtime.h>

// ---------------- types / helpers ----------------
typedef __attribute__((ext_vector_type(16))) __bf16 v16bf;
typedef __attribute__((ext_vector_type(8)))  float  v8f;

#if __has_builtin(__builtin_amdgcn_global_load_async_to_lds_b128) && \
    __has_builtin(__builtin_amdgcn_s_wait_asynccnt)
#define HAVE_ASYNC_LDS 1
#else
#define HAVE_ASYNC_LDS 0
#endif

// exact param type from hipcc diagnostic: pointer to int __vector(4), pointee in AS1/AS3
typedef __attribute__((__vector_size__(4 * sizeof(int)))) int i32x4;
typedef __attribute__((address_space(1))) i32x4* gptr128;
typedef __attribute__((address_space(3))) i32x4* lptr128;

__device__ __forceinline__ void async_copy_b128(const void* g, void* l) {
#if HAVE_ASYNC_LDS
    __builtin_amdgcn_global_load_async_to_lds_b128((gptr128)g, (lptr128)l, 0, 0);
#endif
}

__device__ __forceinline__ unsigned short f2bf(float x) {
    union { float f; unsigned u; } a; a.f = x;
    unsigned u = a.u;
    unsigned r = (u + 0x7FFFu + ((u >> 16) & 1u)) >> 16;   // RNE fp32->bf16
    return (unsigned short)r;
}
__device__ __forceinline__ float bf2f(unsigned short b) {
    union { unsigned u; float f; } a; a.u = ((unsigned)b) << 16; return a.f;
}

// ---------------- model constants ----------------
#define T_SEQ 2048
#define HID   1024
#define NHEAD 16
#define HDIM  64
#define FFD   4096
#define NEXP  8
#define NLAY  2
#define VOC   32000

// ---------------- tiled bf16 WMMA GEMM ----------------
// C[M,N] = A[M,K] (bf16 row-major, lda) @ B (bf16)
//   BNT=1: B stored [N,K] row-major (x @ W.T)
//   BNT=0: B stored [K,N] row-major (x @ W), transposed during LDS fill
// Double-buffered LDS; global->LDS via async DMA (ASYNCcnt) when available.
// Epilogue: +bias[n], exact gelu, *rowScale[m*rsStride], fp32 (opt +=) and/or bf16 out.
#define BM 128
#define BN 128
#define BK 32

template <int BNT>
__launch_bounds__(256)
__global__ void k_gemm(const unsigned short* __restrict__ A, int lda,
                       const unsigned short* __restrict__ B, int ldb,
                       int M, int N, int K,
                       const float* __restrict__ bias,
                       const float* __restrict__ rowScale, int rsStride,
                       float* __restrict__ outF, int accum,
                       unsigned short* __restrict__ outB, int ldc, int doGelu)
{
    __shared__ unsigned short As[2][BM][BK + 8];   // row stride 80B (16B aligned)
    __shared__ unsigned short Bs[2][BN][BK + 8];

    const int tid  = threadIdx.x;
    const int m0   = blockIdx.y * BM;
    const int n0   = blockIdx.x * BN;
    const int wave = tid >> 5;
    const int lane = tid & 31;
    const int wm   = wave >> 2;        // 0..1 -> 64 rows
    const int wn   = wave & 3;         // 0..3 -> 32 cols
    const int lrow = lane & 15;
    const int kbase = (lane >> 4) * 8; // ISA 16-bit layout: hi half-wave holds K+8

    // cooperative-load coordinates
    const int lr   = tid >> 1, lseg = (tid & 1) * 16;  // A and B-NT: 2 thr/row, 16 elems
    const int kk   = tid >> 3, nseg = (tid & 7) * 16;  // B-NN scatter-transpose

    v8f acc[4][2];
    for (int i = 0; i < 4; i++)
        for (int j = 0; j < 2; j++)
            acc[i][j] = (v8f){0.f,0.f,0.f,0.f,0.f,0.f,0.f,0.f};

    union Frag { v16bf v; unsigned u32[8]; };

    auto loadTile = [&](int k0, int buf) {
        // ---- A tile ----
        {
            int m = m0 + lr;
            unsigned short* dst = &As[buf][lr][lseg];
            if (m < M) {
                const unsigned short* src = A + (size_t)m * lda + k0 + lseg;
#if HAVE_ASYNC_LDS
                async_copy_b128(src,     dst);
                async_copy_b128(src + 8, dst + 8);
#else
                const uint4* s4 = (const uint4*)src;
                uint4 a = s4[0], b4 = s4[1];
                unsigned* d = (unsigned*)dst;
                d[0]=a.x; d[1]=a.y; d[2]=a.z; d[3]=a.w;
                d[4]=b4.x; d[5]=b4.y; d[6]=b4.z; d[7]=b4.w;
#endif
            } else {
                unsigned* d = (unsigned*)dst;
                for (int i = 0; i < 8; i++) d[i] = 0u;
            }
        }
        // ---- B tile -> Bs[n][k] ----
        if (BNT) {
            int n = n0 + lr;
            unsigned short* dst = &Bs[buf][lr][lseg];
            if (n < N) {
                const unsigned short* src = B + (size_t)n * ldb + k0 + lseg;
#if HAVE_ASYNC_LDS
                async_copy_b128(src,     dst);
                async_copy_b128(src + 8, dst + 8);
#else
                const uint4* s4 = (const uint4*)src;
                uint4 a = s4[0], b4 = s4[1];
                unsigned* d = (unsigned*)dst;
                d[0]=a.x; d[1]=a.y; d[2]=a.z; d[3]=a.w;
                d[4]=b4.x; d[5]=b4.y; d[6]=b4.z; d[7]=b4.w;
#endif
            } else {
                unsigned* d = (unsigned*)dst;
                for (int i = 0; i < 8; i++) d[i] = 0u;
            }
        } else {
            // B is [K,N]: read 16 contiguous n at row k, transpose into LDS
            if (n0 + nseg < N) {
                const unsigned short* src = B + (size_t)(k0 + kk) * ldb + n0 + nseg;
                #pragma unroll
                for (int i = 0; i < 16; i++) Bs[buf][nseg + i][kk] = src[i];
            } else {
                #pragma unroll
                for (int i = 0; i < 16; i++) Bs[buf][nseg + i][kk] = 0;
            }
        }
    };

    // async loads per tile issued by this thread (for counted asynccnt waits)
    constexpr unsigned short PER_TILE = BNT ? 4 : 2;

    const int nk = K / BK;     // K is always a multiple of 32 in this model
    loadTile(0, 0);

    for (int i = 0; i < nk; i++) {
        __syncthreads();                    // buf[(i+1)&1] no longer being read
        if (i + 1 < nk) {
            loadTile((i + 1) * BK, (i + 1) & 1);
#if HAVE_ASYNC_LDS
            __builtin_amdgcn_s_wait_asynccnt(PER_TILE);   // tile i resident (in-order)
#endif
        } else {
#if HAVE_ASYNC_LDS
            __builtin_amdgcn_s_wait_asynccnt(0);
#endif
        }
        __syncthreads();                    // tile i visible to every wave

        const int buf = i & 1;
        Frag af[4], bfr[2];
        #pragma unroll
        for (int mi = 0; mi < 4; mi++) {
            int mr = wm * 64 + mi * 16 + lrow;
            #pragma unroll
            for (int hh = 0; hh < 2; hh++)
                #pragma unroll
                for (int j = 0; j < 4; j++)
                    af[mi].u32[hh * 4 + j] =
                        *(const unsigned*)&As[buf][mr][hh * 16 + kbase + 2 * j];
        }
        #pragma unroll
        for (int ni = 0; ni < 2; ni++) {
            int nr = wn * 32 + ni * 16 + lrow;
            #pragma unroll
            for (int hh = 0; hh < 2; hh++)
                #pragma unroll
                for (int j = 0; j < 4; j++)
                    bfr[ni].u32[hh * 4 + j] =
                        *(const unsigned*)&Bs[buf][nr][hh * 16 + kbase + 2 * j];
        }
        #pragma unroll
        for (int mi = 0; mi < 4; mi++)
            #pragma unroll
            for (int ni = 0; ni < 2; ni++)
                acc[mi][ni] = __builtin_amdgcn_wmma_f32_16x16x32_bf16(
                    false, af[mi].v, false, bfr[ni].v,
                    (short)0, acc[mi][ni], false, false);
    }

    // ---- epilogue (32-bit C/D layout: n = lane&15, row r + 8*(lane>=16)) ----
    const int rowAdd = (lane >> 4) * 8;
    const int ncol   = lane & 15;
    for (int mi = 0; mi < 4; mi++) {
        for (int ni = 0; ni < 2; ni++) {
            #pragma unroll
            for (int r = 0; r < 8; r++) {
                int m = m0 + wm * 64 + mi * 16 + r + rowAdd;
                int n = n0 + wn * 32 + ni * 16 + ncol;
                if (m < M && n < N) {
                    float v = acc[mi][ni][r];
                    if (bias)     v += bias[n];
                    if (doGelu)   v = 0.5f * v * (1.0f + erff(v * 0.70710678118654752f));
                    if (rowScale) v *= rowScale[(size_t)m * rsStride];
                    size_t o = (size_t)m * ldc + n;
                    if (outF) { if (accum) outF[o] += v; else outF[o] = v; }
                    if (outB) outB[o] = f2bf(v);
                }
            }
        }
    }
}

// ---------------- pointwise / reduction kernels ----------------
__global__ void k_embed(const int* __restrict__ idx, const float* __restrict__ tok,
                        const float* __restrict__ pos, float* __restrict__ x) {
    int i = blockIdx.x * 256 + threadIdx.x;
    if (i >= T_SEQ * HID) return;
    int t = i / HID, h = i % HID;
    x[i] = tok[(size_t)idx[t] * HID + h] + pos[(size_t)t * HID + h];
}

__global__ void k_convert(const float* __restrict__ in, unsigned short* __restrict__ out,
                          size_t n) {
    size_t i = (size_t)blockIdx.x * 256 + threadIdx.x;
    if (i < n) out[i] = f2bf(in[i]);
}

__global__ void k_layernorm(const float* __restrict__ x, const float* __restrict__ g,
                            const float* __restrict__ b, unsigned short* __restrict__ out) {
    int row = blockIdx.x;
    const float* xr = x + (size_t)row * HID;
    __shared__ float red[256];
    float s = 0.f;
    for (int h = threadIdx.x; h < HID; h += 256) s += xr[h];
    red[threadIdx.x] = s; __syncthreads();
    for (int o = 128; o > 0; o >>= 1) {
        if (threadIdx.x < o) red[threadIdx.x] += red[threadIdx.x + o];
        __syncthreads();
    }
    float mu = red[0] * (1.0f / HID);
    __syncthreads();
    float v = 0.f;
    for (int h = threadIdx.x; h < HID; h += 256) { float d = xr[h] - mu; v += d * d; }
    red[threadIdx.x] = v; __syncthreads();
    for (int o = 128; o > 0; o >>= 1) {
        if (threadIdx.x < o) red[threadIdx.x] += red[threadIdx.x + o];
        __syncthreads();
    }
    float rstd = rsqrtf(red[0] * (1.0f / HID) + 1e-5f);
    for (int h = threadIdx.x; h < HID; h += 256)
        out[(size_t)row * HID + h] = f2bf((xr[h] - mu) * rstd * g[h] + b[h]);
}

__global__ void k_split_qkv(const float* __restrict__ qkv,
                            unsigned short* __restrict__ qh,
                            unsigned short* __restrict__ kh,
                            unsigned short* __restrict__ vh) {
    int i = blockIdx.x * 256 + threadIdx.x;
    if (i >= NHEAD * T_SEQ * HDIM) return;
    int h = i / (T_SEQ * HDIM);
    int r = i % (T_SEQ * HDIM);
    int t = r / HDIM, d = r % HDIM;
    size_t src = (size_t)t * 3 * HID + (size_t)h * HDIM + d;
    qh[i] = f2bf(qkv[src]);
    kh[i] = f2bf(qkv[src + HID]);
    vh[i] = f2bf(qkv[src + 2 * HID]);
}

// causal softmax over one score row; writes bf16 probs (zeros above diagonal)
__global__ void k_softmax(const float* __restrict__ scores,
                          unsigned short* __restrict__ probs, float scale) {
    int t = blockIdx.x;
    const float* sr = scores + (size_t)t * T_SEQ;
    unsigned short* pr = probs + (size_t)t * T_SEQ;
    __shared__ float red[256];
    float m = -1e30f;
    for (int j = threadIdx.x; j <= t; j += 256) m = fmaxf(m, sr[j] * scale);
    red[threadIdx.x] = m; __syncthreads();
    for (int o = 128; o > 0; o >>= 1) {
        if (threadIdx.x < o) red[threadIdx.x] = fmaxf(red[threadIdx.x], red[threadIdx.x + o]);
        __syncthreads();
    }
    m = red[0]; __syncthreads();
    float s = 0.f;
    for (int j = threadIdx.x; j <= t; j += 256) s += __expf(sr[j] * scale - m);
    red[threadIdx.x] = s; __syncthreads();
    for (int o = 128; o > 0; o >>= 1) {
        if (threadIdx.x < o) red[threadIdx.x] += red[threadIdx.x + o];
        __syncthreads();
    }
    float inv = 1.0f / red[0];
    for (int j = threadIdx.x; j < T_SEQ; j += 256)
        pr[j] = (j <= t) ? f2bf(__expf(sr[j] * scale - m) * inv) : (unsigned short)0;
}

__global__ void k_merge(const float* __restrict__ headout,
                        unsigned short* __restrict__ attnb) {
    int i = blockIdx.x * 256 + threadIdx.x;
    if (i >= T_SEQ * HID) return;
    int t = i / HID, c = i % HID;
    int h = c / HDIM, d = c % HDIM;
    attnb[i] = f2bf(headout[((size_t)h * T_SEQ + t) * HDIM + d]);
}

// router: softmax over E=8 logits, keep top-2 probs, zeros elsewhere
__global__ void k_router(const unsigned short* __restrict__ xb,
                         const float* __restrict__ rw,
                         float* __restrict__ coef) {
    int wave = threadIdx.x >> 5, lane = threadIdx.x & 31;
    int t = blockIdx.x * 8 + wave;
    if (t >= T_SEQ) return;
    const unsigned short* xr = xb + (size_t)t * HID;
    float lg[NEXP];
    for (int e = 0; e < NEXP; e++) {
        const float* w = rw + (size_t)e * HID;
        float s = 0.f;
        for (int h = lane; h < HID; h += 32) s += bf2f(xr[h]) * w[h];
        for (int o = 16; o > 0; o >>= 1) s += __shfl_xor(s, o, 32);
        lg[e] = s;
    }
    float m = lg[0];
    for (int e = 1; e < NEXP; e++) m = fmaxf(m, lg[e]);
    float p[NEXP], sum = 0.f;
    for (int e = 0; e < NEXP; e++) { p[e] = __expf(lg[e] - m); sum += p[e]; }
    float inv = 1.0f / sum;
    for (int e = 0; e < NEXP; e++) p[e] *= inv;
    int i1 = 0;
    for (int e = 1; e < NEXP; e++) if (p[e] > p[i1]) i1 = e;
    int i2 = -1;
    for (int e = 0; e < NEXP; e++) { if (e == i1) continue; if (i2 < 0 || p[e] > p[i2]) i2 = e; }
    if (lane < NEXP)
        coef[(size_t)t * NEXP + lane] = (lane == i1 || lane == i2) ? p[lane] : 0.0f;
}

// ---------------- host orchestration ----------------
static inline void gemm(hipStream_t s,
                        const unsigned short* A, int lda,
                        const unsigned short* B, int ldb, int bNT,
                        int M, int N, int K,
                        const float* bias, const float* rowScale, int rsStride,
                        float* outF, int accum,
                        unsigned short* outB, int ldc, int doGelu) {
    dim3 g((N + BN - 1) / BN, (M + BM - 1) / BM), b(256);
    if (bNT)
        k_gemm<1><<<g, b, 0, s>>>(A, lda, B, ldb, M, N, K,
                                  bias, rowScale, rsStride, outF, accum, outB, ldc, doGelu);
    else
        k_gemm<0><<<g, b, 0, s>>>(A, lda, B, ldb, M, N, K,
                                  bias, rowScale, rsStride, outF, accum, outB, ldc, doGelu);
}

extern "C" void kernel_launch(void* const* d_in, const int* in_sizes, int n_in,
                              void* d_out, int out_size, void* d_ws, size_t ws_size,
                              hipStream_t stream) {
    const int*   idx   = (const int*)  d_in[0];
    const float* tok   = (const float*)d_in[1];
    const float* pos   = (const float*)d_in[2];
    const float* ln1g  = (const float*)d_in[3];
    const float* ln1b  = (const float*)d_in[4];
    const float* wqkv  = (const float*)d_in[5];
    const float* bqkv  = (const float*)d_in[6];
    const float* wo    = (const float*)d_in[7];
    const float* bo    = (const float*)d_in[8];
    const float* ln2g  = (const float*)d_in[9];
    const float* ln2b  = (const float*)d_in[10];
    const float* rw    = (const float*)d_in[11];
    const float* w1    = (const float*)d_in[12];
    const float* w2    = (const float*)d_in[13];
    const float* lnfg  = (const float*)d_in[14];
    const float* lnfb  = (const float*)d_in[15];
    float* outp = (float*)d_out;

    char* ws = (char*)d_ws;
    size_t off = 0;
    auto carve = [&](size_t bytes) -> char* {
        char* p = ws + off;
        off += (bytes + 255) & ~(size_t)255;
        return p;
    };
    float*          x       = (float*)         carve((size_t)T_SEQ * HID * 4);
    unsigned short* xb      = (unsigned short*)carve((size_t)T_SEQ * HID * 2);
    float*          qkv     = (float*)         carve((size_t)T_SEQ * 3 * HID * 4);
    unsigned short* qh      = (unsigned short*)carve((size_t)NHEAD * T_SEQ * HDIM * 2);
    unsigned short* kh      = (unsigned short*)carve((size_t)NHEAD * T_SEQ * HDIM * 2);
    unsigned short* vh      = (unsigned short*)carve((size_t)NHEAD * T_SEQ * HDIM * 2);
    float*          scores  = (float*)         carve((size_t)T_SEQ * T_SEQ * 4);
    unsigned short* probs   = (unsigned short*)carve((size_t)T_SEQ * T_SEQ * 2);
    float*          headout = (float*)         carve((size_t)NHEAD * T_SEQ * HDIM * 4);
    unsigned short* attnb   = (unsigned short*)carve((size_t)T_SEQ * HID * 2);
    unsigned short* hb      = (unsigned short*)carve((size_t)T_SEQ * FFD * 2);
    unsigned short* wb      = (unsigned short*)carve((size_t)3 * HID * HID * 2);
    unsigned short* w1b     = (unsigned short*)carve((size_t)HID * FFD * 2);
    unsigned short* w2b     = (unsigned short*)carve((size_t)FFD * HID * 2);
    float*          coef    = (float*)         carve((size_t)T_SEQ * NEXP * 4);
    unsigned short* tokb    = (unsigned short*)carve((size_t)VOC * HID * 2);

    auto cvt = [&](const float* in, unsigned short* out, size_t n) {
        k_convert<<<(unsigned)((n + 255) / 256), 256, 0, stream>>>(in, out, n);
    };

    // x = tok_emb[idx] + pos_emb
    k_embed<<<(T_SEQ * HID) / 256, 256, 0, stream>>>(idx, tok, pos, x);
    // bf16 tied lm_head weights (used once at the end)
    cvt(tok, tokb, (size_t)VOC * HID);

    for (int l = 0; l < NLAY; l++) {
        // ---- attention ----
        k_layernorm<<<T_SEQ, 256, 0, stream>>>(x, ln1g + l * HID, ln1b + l * HID, xb);
        cvt(wqkv + (size_t)l * 3 * HID * HID, wb, (size_t)3 * HID * HID);
        gemm(stream, xb, HID, wb, HID, 1, T_SEQ, 3 * HID, HID,
             bqkv + (size_t)l * 3 * HID, nullptr, 0, qkv, 0, nullptr, 3 * HID, 0);
        k_split_qkv<<<(NHEAD * T_SEQ * HDIM) / 256, 256, 0, stream>>>(qkv, qh, kh, vh);
        for (int h = 0; h < NHEAD; h++) {
            // scores = q @ k^T   (NT, K=64)
            gemm(stream, qh + (size_t)h * T_SEQ * HDIM, HDIM,
                 kh + (size_t)h * T_SEQ * HDIM, HDIM, 1,
                 T_SEQ, T_SEQ, HDIM, nullptr, nullptr, 0, scores, 0, nullptr, T_SEQ, 0);
            k_softmax<<<T_SEQ, 256, 0, stream>>>(scores, probs, 0.125f); // 1/sqrt(64)
            // head_out = probs @ v  (NN, N=64)
            gemm(stream, probs, T_SEQ,
                 vh + (size_t)h * T_SEQ * HDIM, HDIM, 0,
                 T_SEQ, HDIM, T_SEQ, nullptr, nullptr, 0,
                 headout + (size_t)h * T_SEQ * HDIM, 0, nullptr, HDIM, 0);
        }
        k_merge<<<(T_SEQ * HID) / 256, 256, 0, stream>>>(headout, attnb);
        cvt(wo + (size_t)l * HID * HID, wb, (size_t)HID * HID);
        // x += attn @ wo^T + bo
        gemm(stream, attnb, HID, wb, HID, 1, T_SEQ, HID, HID,
             bo + (size_t)l * HID, nullptr, 0, x, 1, nullptr, HID, 0);

        // ---- MoE ----
        k_layernorm<<<T_SEQ, 256, 0, stream>>>(x, ln2g + l * HID, ln2b + l * HID, xb);
        k_router<<<T_SEQ / 8, 256, 0, stream>>>(xb, rw + (size_t)l * NEXP * HID, coef);
        for (int e = 0; e < NEXP; e++) {
            cvt(w1 + ((size_t)l * NEXP + e) * HID * FFD, w1b, (size_t)HID * FFD);
            // hb = gelu(xb @ w1)  (NN), bf16 out
            gemm(stream, xb, HID, w1b, FFD, 0, T_SEQ, FFD, HID,
                 nullptr, nullptr, 0, nullptr, 0, hb, FFD, 1);
            cvt(w2 + ((size_t)l * NEXP + e) * FFD * HID, w2b, (size_t)FFD * HID);
            // x += coef[:,e] * (hb @ w2)  (NN)
            gemm(stream, hb, FFD, w2b, HID, 0, T_SEQ, HID, FFD,
                 nullptr, coef + e, NEXP, x, 1, nullptr, HID, 0);
        }
    }

    // final LN + tied lm_head
    k_layernorm<<<T_SEQ, 256, 0, stream>>>(x, lnfg, lnfb, xb);
    gemm(stream, xb, HID, tokb, HID, 1, T_SEQ, VOC, HID,
         nullptr, nullptr, 0, outp, 0, nullptr, VOC, 0);
}
